// FactorAttention_38680475468485
// MI455X (gfx1250) — compile-verified, matrix operand order checked by reading, not run on hardware
//
#include <hip/hip_runtime.h>
#include <hip/hip_bf16.h>

#define D_MODEL  1024
#define NHEAD    16
#define HEAD_DIM 64
#define SEQ      2048
#define BATCH    4
#define NTOK     (BATCH * SEQ)   /* 8192 token rows */

typedef __attribute__((ext_vector_type(16))) _Float16 v16h;
typedef __attribute__((ext_vector_type(8)))  _Float16 v8h;
typedef __attribute__((ext_vector_type(4)))  _Float16 v4h;
typedef __attribute__((ext_vector_type(8)))  float    v8f;
typedef __attribute__((ext_vector_type(4)))  float    v4f;

#define WMMA_F16(A, B, C) \
  __builtin_amdgcn_wmma_f32_16x16x32_f16(false, (A), false, (B), (short)0, (C), false, false)

// Build a 16x32 f16 WMMA operand fragment from a pointer already offset by the
// per-lane element offset (lane l<16: row l, k={0..7,16..23}; lane l>=16:
// row l-16, k={8..15,24..31}).  Works for global and LDS pointers.
static __device__ inline v16h frag_from(const _Float16* p) {
  v8h lo = *(const v8h*)(p);
  v8h hi = *(const v8h*)(p + 16);
  v16h f;
#pragma unroll
  for (int i = 0; i < 8; ++i) { f[i] = lo[i]; f[i + 8] = hi[i]; }
  return f;
}

// ---------------------------------------------------------------------------
// fp32 -> f16 conversion, 4 elements / thread (n divisible by 4)
// ---------------------------------------------------------------------------
__global__ void conv_f32_to_f16(const float* __restrict__ in,
                                _Float16* __restrict__ out, int n4) {
  int i = blockIdx.x * blockDim.x + threadIdx.x;
  if (i >= n4) return;
  v4f v = *(const v4f*)(in + (size_t)i * 4);
  v4h o;
#pragma unroll
  for (int j = 0; j < 4; ++j) o[j] = (_Float16)v[j];
  *(v4h*)(out + (size_t)i * 4) = o;
}

// ---------------------------------------------------------------------------
// WMMA GEMM: C[n,m] = sum_k A[n,k] * W[m,k] + bias[m]
// A: [N x K] f16 row-major, W: [M x K] f16 row-major.
// Block = 8 waves, each computing a 32x64 tile (8 wmma / 32-wide k-step).
// The 32x32 A strip (shared by all 8 waves) is double-buffered in LDS via
// GLOBAL_LOAD_ASYNC_TO_LDS_B64 (256 threads x 8B = 2KB tile), overlapped with
// the WMMA work on the previous strip.  W fragments are direct VMEM b128
// loads with a wave-uniform scalar base (saddr form).
// ---------------------------------------------------------------------------
__global__ __launch_bounds__(256) void gemm_wmma(
    const _Float16* __restrict__ A, const _Float16* __restrict__ W,
    const float* __restrict__ bias, float* __restrict__ Cf,
    _Float16* __restrict__ Ch, int N, int M, int K) {
  __shared__ _Float16 ldsA[2][32][32];

  const int tid  = threadIdx.x;
  const int wave = tid >> 5;
  const int m0   = blockIdx.x * 512 + wave * 64;
  const int n0   = blockIdx.y * 32;
  const int lane = tid & 31;
  const int col  = lane & 15;
  const int hf   = lane >> 4;

  // async producer mapping: thread copies 8B; 256 threads cover 32x32 halves
  const int arow = tid >> 3;
  const int acol = (tid & 7) * 4;  // halves
  const uint32_t lds0 = (uint32_t)(uintptr_t)(&ldsA[0][0][0]);
  const uint32_t lds1 = (uint32_t)(uintptr_t)(&ldsA[1][0][0]);
  const uint32_t ldst = (uint32_t)((arow * 32 + acol) * 2);  // bytes in tile
  const uint64_t grow = (uint64_t)(uintptr_t)(A + (size_t)(n0 + arow) * K + acol);

  // loop-invariant per-lane fragment offsets (elements)
  const uint32_t a0off = (uint32_t)(col * 32 + hf * 8);
  const uint32_t a1off = a0off + 16 * 32;
  const uint32_t wloff = (uint32_t)(col * K + hf * 8);

  v8f acc[2][4];
#pragma unroll
  for (int r = 0; r < 2; ++r)
#pragma unroll
    for (int t = 0; t < 4; ++t) acc[r][t] = (v8f)(0.0f);

  // prologue: stage k=0 strip into buffer 0
  {
    uint32_t la = lds0 + ldst;
    uint64_t ga = grow;
    asm volatile("global_load_async_to_lds_b64 %0, %1, off"
                 :: "v"(la), "v"(ga) : "memory");
    asm volatile("s_wait_asynccnt 0x0" ::: "memory");
  }
  __syncthreads();

  for (int ks = 0; ks < K; ks += 32) {
    const int pb = (ks >> 5) & 1;
    // prefetch next A strip into the other buffer (readers of it finished
    // one barrier ago), overlapped with this step's WMMAs
    if (ks + 32 < K) {
      uint32_t la = (pb ? lds0 : lds1) + ldst;
      uint64_t ga = grow + (uint64_t)(ks + 32) * sizeof(_Float16);
      asm volatile("global_load_async_to_lds_b64 %0, %1, off"
                   :: "v"(la), "v"(ga) : "memory");
    }

    const _Float16* abase = &ldsA[pb][0][0];
    v16h a0 = frag_from(abase + a0off);
    v16h a1 = frag_from(abase + a1off);

    const _Float16* wb = W + (size_t)m0 * K + ks;  // wave-uniform base
    v16h wf[4];
#pragma unroll
    for (int t = 0; t < 4; ++t)
      wf[t] = frag_from(wb + t * 16 * K + wloff);

#pragma unroll
    for (int t = 0; t < 4; ++t) {
      acc[0][t] = WMMA_F16(a0, wf[t], acc[0][t]);
      acc[1][t] = WMMA_F16(a1, wf[t], acc[1][t]);
    }

    asm volatile("s_wait_asynccnt 0x0" ::: "memory");
    __syncthreads();
  }

#pragma unroll
  for (int r = 0; r < 2; ++r)
#pragma unroll
    for (int t = 0; t < 4; ++t)
#pragma unroll
      for (int i = 0; i < 8; ++i) {
        int row = n0 + r * 16 + i + 8 * hf;
        int cc  = m0 + t * 16 + col;
        float v = acc[r][t][i] + bias[cc];
        if (Ch) Ch[(size_t)row * M + cc] = (_Float16)v;
        else    Cf[(size_t)row * M + cc] = v;
      }
}

// ---------------------------------------------------------------------------
// QK-norm: rows of HEAD_DIM=64 contiguous halves; t = t/max(||t||,eps)*scale
// ---------------------------------------------------------------------------
__global__ void qknorm_kernel(_Float16* __restrict__ T,
                              const float* __restrict__ scale, int nrows) {
  int rid = blockIdx.x * blockDim.x + threadIdx.x;
  if (rid >= nrows) return;
  v8h* p = (v8h*)(T + (size_t)rid * HEAD_DIM);
  v8h d[8];
  float ss = 0.0f;
#pragma unroll
  for (int j = 0; j < 8; ++j) {
    d[j] = p[j];
#pragma unroll
    for (int i = 0; i < 8; ++i) { float v = (float)d[j][i]; ss += v * v; }
  }
  float r = scale[0] / fmaxf(sqrtf(ss), 1e-12f);
#pragma unroll
  for (int j = 0; j < 8; ++j) {
#pragma unroll
    for (int i = 0; i < 8; ++i) d[j][i] = (_Float16)((float)d[j][i] * r);
    p[j] = d[j];
  }
}

// ---------------------------------------------------------------------------
// V transpose per head: Vt[b,h,d,l] = V[(b*L+l)*D + h*64 + d]
// ---------------------------------------------------------------------------
__global__ void vtrans_kernel(const _Float16* __restrict__ V,
                              _Float16* __restrict__ Vt, int total) {
  int o = blockIdx.x * blockDim.x + threadIdx.x;
  if (o >= total) return;
  int l = o & (SEQ - 1);
  int d = (o / SEQ) & (HEAD_DIM - 1);
  int h = (o / (SEQ * HEAD_DIM)) & (NHEAD - 1);
  int b = o / (SEQ * HEAD_DIM * NHEAD);
  Vt[o] = V[((size_t)b * SEQ + l) * D_MODEL + h * HEAD_DIM + d];
}

// ---------------------------------------------------------------------------
// Flash attention, causal.  Block = 4 waves; wave owns 16 query rows of one
// (b,h); streams 32-key blocks.  S via 4 WMMAs, online softmax with width-16
// shuffles, P re-layout through a private 1KB LDS slice, PV via 4 WMMAs.
// ---------------------------------------------------------------------------
__global__ __launch_bounds__(128) void attn_kernel(
    const _Float16* __restrict__ Qh, const _Float16* __restrict__ Kh,
    const _Float16* __restrict__ Vt, _Float16* __restrict__ Ah) {
  __shared__ _Float16 ldsP[4][16][32];

  const int b    = blockIdx.z;
  const int h    = blockIdx.y;
  const int wave = threadIdx.x >> 5;
  const int q0   = blockIdx.x * 64 + wave * 16;
  const int lane = threadIdx.x & 31;
  const int col  = lane & 15;
  const int hf   = lane >> 4;

  // loop-invariant per-lane fragment offsets (elements)
  const uint32_t fo = (uint32_t)(col * D_MODEL + hf * 8);  // Q/K row stride D
  const uint32_t vo = (uint32_t)(col * SEQ + hf * 8);      // Vt row stride L

  const _Float16* qbase = Qh + ((size_t)b * SEQ + q0) * D_MODEL + h * HEAD_DIM;
  v16h qf0 = frag_from(qbase + fo);
  v16h qf1 = frag_from(qbase + 32 + fo);

  v8f o[4];
#pragma unroll
  for (int t = 0; t < 4; ++t) o[t] = (v8f)(0.0f);
  float m[8], l[8];
#pragma unroll
  for (int i = 0; i < 8; ++i) { m[i] = -1e30f; l[i] = 0.0f; }

  const _Float16* kroot = Kh + (size_t)b * SEQ * D_MODEL + h * HEAD_DIM;
  const _Float16* vroot = Vt + (((size_t)b * NHEAD + h) * HEAD_DIM) * SEQ;

  const int jmax = (q0 + 15) >> 5;
  for (int jb = 0; jb <= jmax; ++jb) {
    const int k0 = jb * 32;
    const _Float16* kbase = kroot + (size_t)k0 * D_MODEL;

    v16h kf00 = frag_from(kbase + fo);                     // keys 0-15,  d 0-31
    v16h kf01 = frag_from(kbase + 32 + fo);                // keys 0-15,  d 32-63
    v16h kf10 = frag_from(kbase + 16 * D_MODEL + fo);      // keys 16-31, d 0-31
    v16h kf11 = frag_from(kbase + 16 * D_MODEL + 32 + fo);

    v8f sa = (v8f)(0.0f), sb = (v8f)(0.0f);
    sa = WMMA_F16(qf0, kf00, sa); sa = WMMA_F16(qf1, kf01, sa);
    sb = WMMA_F16(qf0, kf10, sb); sb = WMMA_F16(qf1, kf11, sb);

    // scale by hd^-1/2 and apply causal mask
    const int ka = k0 + col, kb2 = k0 + 16 + col;
#pragma unroll
    for (int i = 0; i < 8; ++i) {
      int q = q0 + i + 8 * hf;
      float va = sa[i] * 0.125f; if (ka  > q) va = -1e30f;
      float vb = sb[i] * 0.125f; if (kb2 > q) vb = -1e30f;
      sa[i] = va; sb[i] = vb;
    }

    // online softmax (row = 16 lanes within a half-wave)
#pragma unroll
    for (int i = 0; i < 8; ++i) {
      float mx = fmaxf(sa[i], sb[i]);
#pragma unroll
      for (int off = 8; off >= 1; off >>= 1)
        mx = fmaxf(mx, __shfl_xor(mx, off, 16));
      float mn    = fmaxf(m[i], mx);
      float alpha = __expf(m[i] - mn);
      float pa = __expf(sa[i] - mn);
      float pb = __expf(sb[i] - mn);
      float rs = pa + pb;
#pragma unroll
      for (int off = 8; off >= 1; off >>= 1)
        rs += __shfl_xor(rs, off, 16);
      l[i] = l[i] * alpha + rs;
      m[i] = mn;
#pragma unroll
      for (int t = 0; t < 4; ++t) o[t][i] *= alpha;
      sa[i] = pa; sb[i] = pb;
    }

    // P through this wave's LDS slice to obtain the A-fragment layout.
    // DS pipe is in-order per wave; asm barrier stops compiler reordering.
#pragma unroll
    for (int i = 0; i < 8; ++i) {
      int row = i + 8 * hf;
      ldsP[wave][row][col]      = (_Float16)sa[i];
      ldsP[wave][row][16 + col] = (_Float16)sb[i];
    }
    asm volatile("s_wait_dscnt 0" ::: "memory");
    v16h pf = frag_from(&ldsP[wave][0][0] + (uint32_t)(col * 32 + hf * 8));

    // PV: B-operand columns are rows of Vt (contiguous along keys)
    const _Float16* vb_ = vroot + k0;
#pragma unroll
    for (int t = 0; t < 4; ++t) {
      v16h vf = frag_from(vb_ + t * 16 * SEQ + vo);
      o[t] = WMMA_F16(pf, vf, o[t]);
    }
  }

  // normalize and store f16 in [B,L,D] layout
#pragma unroll
  for (int i = 0; i < 8; ++i) {
    float inv = 1.0f / l[i];
    int q = q0 + i + 8 * hf;
#pragma unroll
    for (int t = 0; t < 4; ++t) {
      int d = h * HEAD_DIM + t * 16 + col;
      Ah[((size_t)b * SEQ + q) * D_MODEL + d] = (_Float16)(o[t][i] * inv);
    }
  }
}

// ---------------------------------------------------------------------------
extern "C" void kernel_launch(void* const* d_in, const int* in_sizes, int n_in,
                              void* d_out, int out_size, void* d_ws, size_t ws_size,
                              hipStream_t stream) {
  const float* x     = (const float*)d_in[0];
  // d_in[1] = mask (tril) — causality is applied analytically in attn_kernel
  const float* Wq    = (const float*)d_in[2];
  const float* bq    = (const float*)d_in[3];
  const float* Wk    = (const float*)d_in[4];
  const float* bk    = (const float*)d_in[5];
  const float* Wv    = (const float*)d_in[6];
  const float* bv    = (const float*)d_in[7];
  const float* Wo    = (const float*)d_in[8];
  const float* bo    = (const float*)d_in[9];
  const float* scale = (const float*)d_in[10];

  char* wsb = (char*)d_ws;
  const size_t SZ_TOK = (size_t)NTOK * D_MODEL * sizeof(_Float16);    // 16 MB
  const size_t SZ_W   = (size_t)D_MODEL * D_MODEL * sizeof(_Float16); //  2 MB
  _Float16* xh  = (_Float16*)(wsb);
  _Float16* wqh = (_Float16*)(wsb + SZ_TOK);
  _Float16* wkh = (_Float16*)(wsb + SZ_TOK + 1 * SZ_W);
  _Float16* wvh = (_Float16*)(wsb + SZ_TOK + 2 * SZ_W);
  _Float16* woh = (_Float16*)(wsb + SZ_TOK + 3 * SZ_W);
  _Float16* Qh  = (_Float16*)(wsb + SZ_TOK + 4 * SZ_W);
  _Float16* Kh  = (_Float16*)(wsb + 2 * SZ_TOK + 4 * SZ_W);
  _Float16* Vh  = (_Float16*)(wsb + 3 * SZ_TOK + 4 * SZ_W);
  _Float16* Vt  = (_Float16*)(wsb + 4 * SZ_TOK + 4 * SZ_W);
  _Float16* Ah  = (_Float16*)(wsb + 5 * SZ_TOK + 4 * SZ_W);

  const int n_x4 = (NTOK * D_MODEL) / 4;
  const int n_w4 = (D_MODEL * D_MODEL) / 4;
  conv_f32_to_f16<<<(n_x4 + 255) / 256, 256, 0, stream>>>(x,  xh,  n_x4);
  conv_f32_to_f16<<<(n_w4 + 255) / 256, 256, 0, stream>>>(Wq, wqh, n_w4);
  conv_f32_to_f16<<<(n_w4 + 255) / 256, 256, 0, stream>>>(Wk, wkh, n_w4);
  conv_f32_to_f16<<<(n_w4 + 255) / 256, 256, 0, stream>>>(Wv, wvh, n_w4);
  conv_f32_to_f16<<<(n_w4 + 255) / 256, 256, 0, stream>>>(Wo, woh, n_w4);

  dim3 gg(D_MODEL / 512, NTOK / 32);     // (2, 256)
  gemm_wmma<<<gg, 256, 0, stream>>>(xh, wqh, bq, nullptr, Qh, NTOK, D_MODEL, D_MODEL);
  gemm_wmma<<<gg, 256, 0, stream>>>(xh, wkh, bk, nullptr, Kh, NTOK, D_MODEL, D_MODEL);
  gemm_wmma<<<gg, 256, 0, stream>>>(xh, wvh, bv, nullptr, Vh, NTOK, D_MODEL, D_MODEL);

  const int nrows = NTOK * NHEAD;        // 131072
  qknorm_kernel<<<(nrows + 255) / 256, 256, 0, stream>>>(Qh, scale, nrows);
  qknorm_kernel<<<(nrows + 255) / 256, 256, 0, stream>>>(Kh, scale, nrows);

  const int n_v = BATCH * NHEAD * HEAD_DIM * SEQ;
  vtrans_kernel<<<(n_v + 255) / 256, 256, 0, stream>>>(Vh, Vt, n_v);

  dim3 ga(SEQ / 64, NHEAD, BATCH);       // (32, 16, 4), 4 waves/block
  attn_kernel<<<ga, 128, 0, stream>>>(Qh, Kh, Vt, Ah);

  gemm_wmma<<<gg, 256, 0, stream>>>(Ah, woh, bo, (float*)d_out, nullptr,
                                    NTOK, D_MODEL, D_MODEL);
}
